// VectorQuantisizer_57681410785308
// MI455X (gfx1250) — matrix operand order repaired and checked by reading, not compiled, not run on hardware
//
#include <hip/hip_runtime.h>
#include <hip/hip_bf16.h>

typedef __attribute__((ext_vector_type(16))) _Float16 v16h;
typedef __attribute__((ext_vector_type(8)))  float    v8f;

#define EMB_K   512      // codebook entries
#define EMB_D   64       // embedding dim
#define HW      4096     // 64*64 spatial per image
#define TPB     256      // 8 waves (wave32)
#define PTS_PER_BLOCK 128  // 8 waves * 16 points

// out[n, :] = W[argmax_k ( x_n . W_k - 0.5*||W_k||^2 ), :]
__global__ __launch_bounds__(TPB)
void VectorQuantisizer_57681410785308_kernel(const float* __restrict__ x,
                                             const float* __restrict__ Wg,
                                             float* __restrict__ out) {
    // ---- LDS: f16 codebook (64 KB) + 0.5*||w||^2 (2 KB) ----
    __shared__ _Float16 Wl[EMB_K * EMB_D];
    __shared__ float    wn[EMB_K];

    const int tid = threadIdx.x;

    // Stage codebook: thread t converts rows 2t and 2t+1 to f16 and
    // accumulates 0.5*||w||^2 in f32.
    {
        const int r0 = tid * 2;
        #pragma unroll
        for (int rr = 0; rr < 2; ++rr) {
            const int r = r0 + rr;
            float ss = 0.0f;
            #pragma unroll
            for (int k = 0; k < EMB_D; k += 4) {
                float4 v = *(const float4*)(Wg + r * EMB_D + k);
                ss += v.x * v.x + v.y * v.y + v.z * v.z + v.w * v.w;
                Wl[r * EMB_D + k + 0] = (_Float16)v.x;
                Wl[r * EMB_D + k + 1] = (_Float16)v.y;
                Wl[r * EMB_D + k + 2] = (_Float16)v.z;
                Wl[r * EMB_D + k + 3] = (_Float16)v.w;
            }
            wn[r] = 0.5f * ss;
        }
    }
    __syncthreads();

    const int lane  = tid & 31;
    const int wave  = tid >> 5;
    const int mrow  = lane & 15;   // point row within 16x16 tile
    const int kgrp  = lane >> 4;   // which K-half this lane supplies

    // 16 consecutive points handled by this wave
    const int m_base = blockIdx.x * PTS_PER_BLOCK + wave * 16;

    // ---- Build the two A fragments (K = 0..31 and 32..63) ----
    // x layout: [B, C, H, W] -> x(point n, channel c) = x[(n>>12)*262144 + c*4096 + (n&4095)]
    const int npt   = m_base + mrow;
    const long xoff = (long)(npt >> 12) * (EMB_D * HW) + (npt & (HW - 1));

    v16h a0, a1;
    #pragma unroll
    for (int j = 0; j < 8; ++j) {
        // A 16x32 layout (contiguous, matches B): lane half kgrp supplies K = kgrp*16 .. +15
        const int kA = kgrp * 16 + j;
        a0[j]     = (_Float16)x[xoff + (long)(kA)           * HW];
        a0[8 + j] = (_Float16)x[xoff + (long)(kA + 8)       * HW];
        a1[j]     = (_Float16)x[xoff + (long)(kA + 32)      * HW];
        a1[8 + j] = (_Float16)x[xoff + (long)(kA + 40)      * HW];
    }

    // ---- Sweep all 32 code tiles with WMMA, track per-lane running argmax ----
    float best[8];
    int   bidx[8];
    #pragma unroll
    for (int r = 0; r < 8; ++r) { best[r] = -3.402823466e38f; bidx[r] = 0; }

    #pragma unroll 4
    for (int nt = 0; nt < EMB_K / 16; ++nt) {
        const int code = nt * 16 + (lane & 15);
        // B 32x16 f16 layout: lane supplies column `code`, K-rows kgrp*16 .. +15,
        // which is a contiguous 16-half (32 B) slice of the LDS codebook row.
        const v16h b0 = *(const v16h*)(Wl + code * EMB_D + kgrp * 16);
        const v16h b1 = *(const v16h*)(Wl + code * EMB_D + 32 + kgrp * 16);

        v8f acc = {};
        acc = __builtin_amdgcn_wmma_f32_16x16x32_f16(false, a0, false, b0,
                                                     (short)0, acc, false, false);
        acc = __builtin_amdgcn_wmma_f32_16x16x32_f16(false, a1, false, b1,
                                                     (short)0, acc, false, false);

        const float bias = wn[code];
        #pragma unroll
        for (int r = 0; r < 8; ++r) {
            const float s = acc[r] - bias;        // x.w - 0.5||w||^2
            if (s > best[r]) { best[r] = s; bidx[r] = code; }
        }
    }

    // ---- Argmax reduction across the 16 lanes of each half, then gather ----
    #pragma unroll
    for (int r = 0; r < 8; ++r) {
        float b = best[r];
        int   i = bidx[r];
        #pragma unroll
        for (int off = 1; off < 16; off <<= 1) {
            const float ob = __shfl_xor(b, off, 32);
            const int   oi = __shfl_xor(i, off, 32);
            if (ob > b || (ob == b && oi < i)) { b = ob; i = oi; }
        }
        // lanes 0-15 hold winner for row r; lanes 16-31 for row r+8
        const int m  = r + 8 * kgrp;
        const int ng = m_base + m;
        const int c0 = (lane & 15) * 4;           // 16 lanes x float4 = 64 floats
        float4 v = *(const float4*)(Wg  + (long)i  * EMB_D + c0);
        *(float4*)(out + (long)ng * EMB_D + c0) = v;
    }
}

extern "C" void kernel_launch(void* const* d_in, const int* in_sizes, int n_in,
                              void* d_out, int out_size, void* d_ws, size_t ws_size,
                              hipStream_t stream) {
    (void)n_in; (void)d_ws; (void)ws_size; (void)out_size;
    const float* x = (const float*)d_in[0];   // [32, 64, 64, 64] f32, channel-major
    const float* W = (const float*)d_in[1];   // [512, 64] f32
    float* out = (float*)d_out;               // [32, 64, 64, 64] channels-last f32

    const int n_points = in_sizes[0] / EMB_D; // 131072
    const int blocks   = n_points / PTS_PER_BLOCK; // 1024

    VectorQuantisizer_57681410785308_kernel<<<blocks, TPB, 0, stream>>>(x, W, out);
}
